// GATModel_72877005079080
// MI455X (gfx1250) — matrix operand order repaired and checked by reading, not compiled, not run on hardware
//
#include <hip/hip_runtime.h>
#include <hip/hip_bf16.h>
#include <stdint.h>

#define NNODES  100000
#define NEDGES  400000
#define NGRAPHS 4096
#define INDIM   128
#define HIDC    64
#define NHEADS  8
#define TOTE    (NEDGES + NNODES)   // 500000, includes self-loops
#define D1      (NHEADS * HIDC)     // 512

typedef __attribute__((ext_vector_type(16))) _Float16 v16h;
typedef __attribute__((ext_vector_type(8)))  float    v8f;

union HFrag { v16h h; uint32_t u[8]; };

__device__ __forceinline__ unsigned fenc(float f) {
  unsigned u = __float_as_uint(f);
  return (u & 0x80000000u) ? ~u : (u | 0x80000000u);   // order-preserving encode
}
__device__ __forceinline__ float fdec(unsigned u) {
  return (u & 0x80000000u) ? __uint_as_float(u & 0x7fffffffu) : __uint_as_float(~u);
}
__device__ __forceinline__ float lrelu(float a) { return a > 0.f ? a : 0.2f * a; }

__device__ __forceinline__ void edge_sd(int e, const long long* __restrict__ src,
                                        const long long* __restrict__ dst, int& s, int& d) {
  if (e < NEDGES) { s = (int)src[e]; d = (int)dst[e]; }
  else            { s = d = e - NEDGES; }               // self-loops
}

// ---------------- conversions (f32 -> f16, weights transposed) ----------------
__global__ void k_cvt_x(const float* __restrict__ x, _Float16* __restrict__ xh) {
  int i = blockIdx.x * 256 + threadIdx.x;
  if (i < NNODES * INDIM) xh[i] = (_Float16)x[i];
}
__global__ void k_cvt_w1(const float* __restrict__ W1, _Float16* __restrict__ w1T) {
  int i = blockIdx.x * 256 + threadIdx.x;            // i = col*128 + k
  if (i >= D1 * INDIM) return;
  int col = i >> 7, k = i & 127;
  w1T[i] = (_Float16)W1[k * D1 + col];
}
__global__ void k_cvt_w2(const float* __restrict__ W2, _Float16* __restrict__ w2T) {
  int i = blockIdx.x * 256 + threadIdx.x;            // i = col*512 + k
  if (i >= HIDC * D1) return;
  int col = i >> 9, k = i & 511;
  w2T[i] = (_Float16)W2[k * HIDC + col];
}

// ---------------- GEMM1: h1[N,512] = xh[N,128] @ W1 (WMMA f16->f32) ----------------
__global__ void __launch_bounds__(256) k_gemm1(const _Float16* __restrict__ xh,
                                               const _Float16* __restrict__ w1T,
                                               float* __restrict__ h1) {
  const int wave = threadIdx.x >> 5, lane = threadIdx.x & 31;
  const int lm = lane & 15, lhi = lane >> 4;
  const int row0 = blockIdx.x * 16;
  const int row  = row0 + lm;
  v8f acc[4];
#pragma unroll
  for (int t = 0; t < 4; ++t)
#pragma unroll
    for (int r = 0; r < 8; ++r) acc[t][r] = 0.f;
#pragma unroll
  for (int kb = 0; kb < INDIM; kb += 32) {
    HFrag a;
    const uint32_t* ap = (const uint32_t*)(xh + (size_t)row * INDIM + kb);
#pragma unroll
    for (int v = 0; v < 8; ++v) {
      const int k = 2 * v + 16 * lhi;  // contiguous K per lane, matches B
      a.u[v] = ap[k >> 1];
    }
#pragma unroll
    for (int t = 0; t < 4; ++t) {
      const int col = wave * 16 + t * 128 + lm;
      HFrag b;   // B 32x16: lanes0-15 K=0..15, lanes16-31 K=16..31 (contiguous per lane)
      const uint32_t* bp = (const uint32_t*)(w1T + (size_t)col * INDIM + kb + 16 * lhi);
#pragma unroll
      for (int v = 0; v < 8; ++v) b.u[v] = bp[v];
      acc[t] = __builtin_amdgcn_wmma_f32_16x16x32_f16(false, a.h, false, b.h,
                                                      (short)0, acc[t], false, false);
    }
  }
#pragma unroll
  for (int t = 0; t < 4; ++t) {
    const int col = wave * 16 + t * 128 + lm;
#pragma unroll
    for (int r = 0; r < 8; ++r)
      h1[(size_t)(row0 + r + 8 * lhi) * D1 + col] = acc[t][r];
  }
}

// ---------------- GEMM2: h2[N,64] = hin[N,512] @ W2 (WMMA) ----------------
__global__ void __launch_bounds__(128) k_gemm2(const _Float16* __restrict__ hin,
                                               const _Float16* __restrict__ w2T,
                                               float* __restrict__ h2) {
  const int wave = threadIdx.x >> 5, lane = threadIdx.x & 31;
  const int lm = lane & 15, lhi = lane >> 4;
  const int row0 = blockIdx.x * 16;
  const int row  = row0 + lm;
  const int col  = wave * 16 + lm;
  v8f acc;
#pragma unroll
  for (int r = 0; r < 8; ++r) acc[r] = 0.f;
  for (int kb = 0; kb < D1; kb += 32) {
    HFrag a, b;
    const uint32_t* ap = (const uint32_t*)(hin + (size_t)row * D1 + kb);
#pragma unroll
    for (int v = 0; v < 8; ++v)
      a.u[v] = ap[(2 * v + 16 * lhi) >> 1];
    const uint32_t* bp = (const uint32_t*)(w2T + (size_t)col * D1 + kb + 16 * lhi);
#pragma unroll
    for (int v = 0; v < 8; ++v) b.u[v] = bp[v];
    acc = __builtin_amdgcn_wmma_f32_16x16x32_f16(false, a.h, false, b.h,
                                                 (short)0, acc, false, false);
  }
#pragma unroll
  for (int r = 0; r < 8; ++r)
    h2[(size_t)(row0 + r + 8 * lhi) * HIDC + col] = acc[r];
}

// ---------------- per-node attention dot products ----------------
__global__ void k_attn1(const float* __restrict__ h1, const float* __restrict__ asrc,
                        const float* __restrict__ adst, float* __restrict__ as1,
                        float* __restrict__ ad1) {
  const int node = blockIdx.x * 8 + (threadIdx.x >> 5);
  const int lane = threadIdx.x & 31;
  if (node >= NNODES) return;
  const int hh = lane >> 2, q = lane & 3;   // 8 heads x 4 lanes, 16 channels each
  const float* hp = h1 + (size_t)node * D1 + hh * HIDC + q * 16;
  const float* sp = asrc + hh * HIDC + q * 16;
  const float* dp = adst + hh * HIDC + q * 16;
  float ps = 0.f, pd = 0.f;
#pragma unroll
  for (int c = 0; c < 16; ++c) { const float v = hp[c]; ps += v * sp[c]; pd += v * dp[c]; }
  ps += __shfl_xor(ps, 1, 4); ps += __shfl_xor(ps, 2, 4);
  pd += __shfl_xor(pd, 1, 4); pd += __shfl_xor(pd, 2, 4);
  if (q == 0) { as1[node * NHEADS + hh] = ps; ad1[node * NHEADS + hh] = pd; }
}

__global__ void k_attn2(const float* __restrict__ h2, const float* __restrict__ asrc,
                        const float* __restrict__ adst, float* __restrict__ as2,
                        float* __restrict__ ad2) {
  const int node = blockIdx.x * 8 + (threadIdx.x >> 5);
  const int lane = threadIdx.x & 31;
  if (node >= NNODES) return;
  const float* hp = h2 + (size_t)node * HIDC;
  float ps = hp[lane] * asrc[lane] + hp[lane + 32] * asrc[lane + 32];
  float pd = hp[lane] * adst[lane] + hp[lane + 32] * adst[lane + 32];
#pragma unroll
  for (int o = 16; o > 0; o >>= 1) { ps += __shfl_xor(ps, o, 32); pd += __shfl_xor(pd, o, 32); }
  if (lane == 0) { as2[node] = ps; ad2[node] = pd; }
}

// ---------------- layer-1 edge passes (8 heads) ----------------
__global__ void k_emax1(const long long* __restrict__ src, const long long* __restrict__ dst,
                        const float* __restrict__ as1, const float* __restrict__ ad1,
                        unsigned* __restrict__ amax) {
  const int idx = blockIdx.x * 256 + threadIdx.x;
  if (idx >= TOTE * NHEADS) return;
  const int e = idx >> 3, h = idx & 7;
  int s, d; edge_sd(e, src, dst, s, d);
  const float a = lrelu(as1[(size_t)s * NHEADS + h] + ad1[(size_t)d * NHEADS + h]);
  atomicMax(amax + (size_t)d * NHEADS + h, fenc(a));
}
__global__ void k_eexp1(const long long* __restrict__ src, const long long* __restrict__ dst,
                        const float* __restrict__ as1, const float* __restrict__ ad1,
                        const unsigned* __restrict__ amax, float* __restrict__ alphaE,
                        float* __restrict__ denom) {
  const int idx = blockIdx.x * 256 + threadIdx.x;
  if (idx >= TOTE * NHEADS) return;
  const int e = idx >> 3, h = idx & 7;
  int s, d; edge_sd(e, src, dst, s, d);
  const float a = lrelu(as1[(size_t)s * NHEADS + h] + ad1[(size_t)d * NHEADS + h]);
  const float ex = expf(a - fdec(amax[(size_t)d * NHEADS + h]));
  alphaE[idx] = ex;
  atomicAdd(denom + (size_t)d * NHEADS + h, ex);
}
__global__ void k_agg1(const long long* __restrict__ src, const long long* __restrict__ dst,
                       const float* __restrict__ alphaE, const float* __restrict__ denom,
                       const float* __restrict__ h1, float* __restrict__ out1) {
  const int idx = blockIdx.x * 256 + threadIdx.x;      // TOTE*128 threads, 4 ch each
  if (idx >= TOTE * 128) return;
  const int e = idx >> 7;
  const int c = (idx & 127) << 2;
  const int h = c >> 6;
  int s, d; edge_sd(e, src, dst, s, d);
  const float w = alphaE[(size_t)e * NHEADS + h] / (denom[(size_t)d * NHEADS + h] + 1e-16f);
  const float4 hv = *(const float4*)(h1 + (size_t)s * D1 + c);
  float* op = out1 + (size_t)d * D1 + c;
  atomicAdd(op + 0, hv.x * w);
  atomicAdd(op + 1, hv.y * w);
  atomicAdd(op + 2, hv.z * w);
  atomicAdd(op + 3, hv.w * w);
}

// out1 + b1, relu, convert to f16 for GEMM2
__global__ void k_brc(const float* __restrict__ out1, const float* __restrict__ b1,
                      _Float16* __restrict__ hin) {
  const int i = blockIdx.x * 256 + threadIdx.x;
  if (i >= NNODES * D1) return;
  float v = out1[i] + b1[i & 511];
  hin[i] = (_Float16)(v > 0.f ? v : 0.f);
}

// ---------------- layer-2 edge passes (1 head) ----------------
__global__ void k_emax2(const long long* __restrict__ src, const long long* __restrict__ dst,
                        const float* __restrict__ as2, const float* __restrict__ ad2,
                        unsigned* __restrict__ amax) {
  const int e = blockIdx.x * 256 + threadIdx.x;
  if (e >= TOTE) return;
  int s, d; edge_sd(e, src, dst, s, d);
  atomicMax(amax + d, fenc(lrelu(as2[s] + ad2[d])));
}
__global__ void k_eexp2(const long long* __restrict__ src, const long long* __restrict__ dst,
                        const float* __restrict__ as2, const float* __restrict__ ad2,
                        const unsigned* __restrict__ amax, float* __restrict__ alphaE,
                        float* __restrict__ denom) {
  const int e = blockIdx.x * 256 + threadIdx.x;
  if (e >= TOTE) return;
  int s, d; edge_sd(e, src, dst, s, d);
  const float ex = expf(lrelu(as2[s] + ad2[d]) - fdec(amax[d]));
  alphaE[e] = ex;
  atomicAdd(denom + d, ex);
}
__global__ void k_agg2(const long long* __restrict__ src, const long long* __restrict__ dst,
                       const float* __restrict__ alphaE, const float* __restrict__ denom,
                       const float* __restrict__ h2, float* __restrict__ out2) {
  const int idx = blockIdx.x * 256 + threadIdx.x;      // TOTE*16 threads, 4 ch each
  if (idx >= TOTE * 16) return;
  const int e = idx >> 4;
  const int c = (idx & 15) << 2;
  int s, d; edge_sd(e, src, dst, s, d);
  const float w = alphaE[e] / (denom[d] + 1e-16f);
  const float4 hv = *(const float4*)(h2 + (size_t)s * HIDC + c);
  float* op = out2 + (size_t)d * HIDC + c;
  atomicAdd(op + 0, hv.x * w);
  atomicAdd(op + 1, hv.y * w);
  atomicAdd(op + 2, hv.z * w);
  atomicAdd(op + 3, hv.w * w);
}

// bias+relu then mean-pool accumulate per graph
__global__ void k_pool(const float* __restrict__ out2, const float* __restrict__ b2,
                       const long long* __restrict__ batch, float* __restrict__ pool,
                       float* __restrict__ counts) {
  const int idx = blockIdx.x * 256 + threadIdx.x;      // NNODES*64
  if (idx >= NNODES * HIDC) return;
  const int node = idx >> 6, c = idx & 63;
  float v = out2[idx] + b2[c];
  v = v > 0.f ? v : 0.f;
  const long long g = batch[node];
  atomicAdd(pool + (size_t)g * HIDC + c, v);
  if (c == 0) atomicAdd(counts + g, 1.0f);
}

__global__ void k_final(const float* __restrict__ pool, const float* __restrict__ counts,
                        const float* __restrict__ Wf, const float* __restrict__ bf,
                        float* __restrict__ out) {
  const int g = blockIdx.x * 8 + (threadIdx.x >> 5);
  const int lane = threadIdx.x & 31;
  if (g >= NGRAPHS) return;
  float cnt = counts[g];
  cnt = cnt < 1.f ? 1.f : cnt;
  const float* pp = pool + (size_t)g * HIDC;
  float p = (pp[lane] * Wf[lane] + pp[lane + 32] * Wf[lane + 32]) / cnt;
#pragma unroll
  for (int o = 16; o > 0; o >>= 1) p += __shfl_xor(p, o, 32);
  if (lane == 0) out[g] = p + bf[0];
}

// ---------------- host launcher ----------------
extern "C" void kernel_launch(void* const* d_in, const int* in_sizes, int n_in,
                              void* d_out, int out_size, void* d_ws, size_t ws_size,
                              hipStream_t stream) {
  (void)in_sizes; (void)n_in; (void)out_size; (void)ws_size;
  const float*     x     = (const float*)d_in[0];
  const long long* ei    = (const long long*)d_in[1];   // int64 [2, E]
  const long long* batch = (const long long*)d_in[2];   // int64 [N]
  const float*     W1    = (const float*)d_in[3];
  const float*     as1w  = (const float*)d_in[4];
  const float*     ad1w  = (const float*)d_in[5];
  const float*     b1    = (const float*)d_in[6];
  const float*     W2    = (const float*)d_in[7];
  const float*     as2w  = (const float*)d_in[8];
  const float*     ad2w  = (const float*)d_in[9];
  const float*     b2    = (const float*)d_in[10];
  const float*     Wf    = (const float*)d_in[11];
  const float*     bf    = (const float*)d_in[12];
  float* out = (float*)d_out;
  const long long* srcp = ei;
  const long long* dstp = ei + NEDGES;

  // workspace arena (deterministic layout, ~495 MB)
  char* ws = (char*)d_ws;
  size_t off = 0;
  auto take = [&](size_t bytes) -> char* {
    char* p = ws + off;
    off += (bytes + 255) & ~size_t(255);
    return p;
  };
  float*     h1      = (float*)take((size_t)NNODES * D1 * 4);     // also aliased as hin(f16)
  float*     out1    = (float*)take((size_t)NNODES * D1 * 4);
  _Float16*  xh      = (_Float16*)take((size_t)NNODES * INDIM * 2); // also aliased as h2(f32)
  _Float16*  w1T     = (_Float16*)take((size_t)D1 * INDIM * 2);
  _Float16*  w2T     = (_Float16*)take((size_t)HIDC * D1 * 2);
  float*     as1     = (float*)take((size_t)NNODES * NHEADS * 4);
  float*     ad1     = (float*)take((size_t)NNODES * NHEADS * 4);
  unsigned*  amax1   = (unsigned*)take((size_t)NNODES * NHEADS * 4);
  float*     denom1  = (float*)take((size_t)NNODES * NHEADS * 4);
  float*     alphaE1 = (float*)take((size_t)TOTE * NHEADS * 4);
  float*     as2     = (float*)take((size_t)NNODES * 4);
  float*     ad2     = (float*)take((size_t)NNODES * 4);
  unsigned*  amax2   = (unsigned*)take((size_t)NNODES * 4);
  float*     denom2  = (float*)take((size_t)NNODES * 4);
  float*     alphaE2 = (float*)take((size_t)TOTE * 4);
  float*     out2    = (float*)take((size_t)NNODES * HIDC * 4);
  float*     pool    = (float*)take((size_t)NGRAPHS * HIDC * 4);
  float*     counts  = (float*)take((size_t)NGRAPHS * 4);
  _Float16*  hin = (_Float16*)h1;  // valid: h1 dead after k_agg1
  float*     h2  = (float*)xh;     // valid: xh dead after k_gemm1 (both 25.6MB)

  // zero accumulators / sentinels (fenc sentinel == 0 orders below every real float)
  hipMemsetAsync(out1,   0, (size_t)NNODES * D1 * 4, stream);
  hipMemsetAsync(amax1,  0, (size_t)NNODES * NHEADS * 4, stream);
  hipMemsetAsync(denom1, 0, (size_t)NNODES * NHEADS * 4, stream);
  hipMemsetAsync(amax2,  0, (size_t)NNODES * 4, stream);
  hipMemsetAsync(denom2, 0, (size_t)NNODES * 4, stream);
  hipMemsetAsync(out2,   0, (size_t)NNODES * HIDC * 4, stream);
  hipMemsetAsync(pool,   0, (size_t)NGRAPHS * HIDC * 4, stream);
  hipMemsetAsync(counts, 0, (size_t)NGRAPHS * 4, stream);

  // conversions
  k_cvt_x <<<(NNODES * INDIM + 255) / 256, 256, 0, stream>>>(x, xh);
  k_cvt_w1<<<(D1 * INDIM + 255) / 256,     256, 0, stream>>>(W1, w1T);
  k_cvt_w2<<<(HIDC * D1 + 255) / 256,      256, 0, stream>>>(W2, w2T);

  // layer 1
  k_gemm1<<<NNODES / 16, 256, 0, stream>>>(xh, w1T, h1);
  k_attn1<<<(NNODES + 7) / 8, 256, 0, stream>>>(h1, as1w, ad1w, as1, ad1);
  k_emax1<<<(TOTE * NHEADS + 255) / 256, 256, 0, stream>>>(srcp, dstp, as1, ad1, amax1);
  k_eexp1<<<(TOTE * NHEADS + 255) / 256, 256, 0, stream>>>(srcp, dstp, as1, ad1, amax1,
                                                           alphaE1, denom1);
  k_agg1 <<<(TOTE * 128 + 255) / 256, 256, 0, stream>>>(srcp, dstp, alphaE1, denom1, h1, out1);
  k_brc  <<<(NNODES * D1 + 255) / 256, 256, 0, stream>>>(out1, b1, hin);

  // layer 2
  k_gemm2<<<NNODES / 16, 128, 0, stream>>>(hin, w2T, h2);
  k_attn2<<<(NNODES + 7) / 8, 256, 0, stream>>>(h2, as2w, ad2w, as2, ad2);
  k_emax2<<<(TOTE + 255) / 256, 256, 0, stream>>>(srcp, dstp, as2, ad2, amax2);
  k_eexp2<<<(TOTE + 255) / 256, 256, 0, stream>>>(srcp, dstp, as2, ad2, amax2, alphaE2, denom2);
  k_agg2 <<<(TOTE * 16 + 255) / 256, 256, 0, stream>>>(srcp, dstp, alphaE2, denom2, h2, out2);

  // pooling + head
  k_pool <<<(NNODES * HIDC + 255) / 256, 256, 0, stream>>>(out2, b2, batch, pool, counts);
  k_final<<<(NGRAPHS + 7) / 8, 256, 0, stream>>>(pool, counts, Wf, bf, out);
}
